// Attention_nn_57707180589242
// MI455X (gfx1250) — matrix operand order repaired and checked by reading, not compiled, not run on hardware
//
#include <hip/hip_runtime.h>
#include <hip/hip_bf16.h>
#include <stdint.h>

typedef _Float16 v8h  __attribute__((ext_vector_type(8)));
typedef _Float16 v16h __attribute__((ext_vector_type(16)));
typedef float    v8f  __attribute__((ext_vector_type(8)));
typedef uint32_t v4u  __attribute__((ext_vector_type(4)));

constexpr int kB   = 128;
constexpr int kC   = 512;
constexpr int kEMB = 512;
constexpr int kHID = 512;
constexpr int kH   = 8;
constexpr int kW   = 64;
constexpr int kL   = kH * kW;          // 512
constexpr int COLS = 34;               // 32 + 2 halo
constexpr int PSTR = 20;               // padded pair-stride (dwords), 16B aligned

// ---------------- kernel 0: K_conv fp32 [E][C][3][3] -> f16 [j][E][C] ----------------
__global__ void k_prep_weights(const float* __restrict__ K_conv, _Float16* __restrict__ A_h) {
    int i = blockIdx.x * blockDim.x + threadIdx.x;          // < 512*512*9
    if (i >= kEMB * kC * 9) return;
    int e   = i / (kC * 9);
    int rem = i - e * (kC * 9);
    int c   = rem / 9;
    int j   = rem - c * 9;
    A_h[((size_t)j * kEMB + e) * kC + c] = (_Float16)K_conv[i];
}

// ---------------- kernel 1: g_em = h @ W_h^T + b_h (fp32, tiny) ----------------
__global__ void k_gem(const float* __restrict__ h, const float* __restrict__ W_h,
                      const float* __restrict__ b_h, float* __restrict__ g_em) {
    const int b = blockIdx.x;
    const int e = threadIdx.x;                              // blockDim = 512
    __shared__ float sh[kHID];
    sh[e] = h[(size_t)b * kHID + e];
    __syncthreads();
    const float4* wr4 = (const float4*)(W_h + (size_t)e * kHID);
    const float4* sh4 = (const float4*)sh;
    float acc = b_h[e];
#pragma unroll 4
    for (int k = 0; k < kHID / 4; ++k) {
        float4 w = wr4[k], x = sh4[k];
        acc += w.x * x.x + w.y * x.y + w.z * x.z + w.w * x.w;
    }
    g_em[(size_t)b * kEMB + e] = acc;
}

// ---------------- kernel 2: fused implicit-GEMM conv + tanh + w_att dot ----------------
// grid = B*16 WGs; WG = 256 thr (8 waves); each WG: one batch, 32 positions (y fixed).
__global__ void __launch_bounds__(256)
k_conv_att(const float* __restrict__ conv_f,
           const _Float16* __restrict__ A_h,
           const float* __restrict__ g_em,
           const float* __restrict__ b_conv,
           const float* __restrict__ w_att,
           const float* __restrict__ b_att,
           float* __restrict__ Escore)
{
    __shared__ uint32_t sB[3 * COLS * PSTR];   // f16x2 pairs: [row][col][pair<16 pad 20]
    __shared__ float    sBG[kEMB];             // b_conv + g_em
    __shared__ float    sWA[kEMB];             // w_att
    __shared__ float    sE[32];                // per-position score partials

    const int tid  = threadIdx.x;
    const int lane = tid & 31;
    const int wave = tid >> 5;
    const int bb   = blockIdx.x >> 4;          // batch
    const int tile = blockIdx.x & 15;          // 16 tiles of 32 positions
    const int y0   = tile >> 1;
    const int x0   = (tile & 1) * 32;
    const int hi   = (lane & 16) ? 1 : 0;
    const int l15  = lane & 15;

    for (int i = tid; i < kEMB; i += 256) {
        sBG[i] = b_conv[i] + g_em[(size_t)bb * kEMB + i];
        sWA[i] = w_att[i];
    }
    if (tid < 32) sE[tid] = 0.0f;

    v8f zero = {0.f, 0.f, 0.f, 0.f, 0.f, 0.f, 0.f, 0.f};
    v8f acc[4][2];
#pragma unroll
    for (int m = 0; m < 4; ++m)
#pragma unroll
        for (int n = 0; n < 2; ++n) acc[m][n] = zero;

    const int erow0 = wave * 64 + l15;

    for (int chunk = 0; chunk < 16; ++chunk) {
        __syncthreads();
        const int cch = chunk * 32;
        // stage 32 channels x 3 rows x 34 cols halo (f16x2-packed), zero-padded OOB
        for (int i = tid; i < 3 * COLS * 16; i += 256) {
            int pair = i / (3 * COLS);
            int rem  = i - pair * (3 * COLS);
            int row  = rem / COLS;
            int col  = rem - row * COLS;
            int yy = y0 + row - 1;
            int xx = x0 + col - 1;
            uint32_t packed = 0u;
            if ((unsigned)yy < (unsigned)kH && (unsigned)xx < (unsigned)kW) {
                const float* p = conv_f +
                    ((((size_t)bb * kC + (cch + 2 * pair)) * kH + yy) * kW + xx);
                union { _Float16 h2[2]; uint32_t u; } pk;
                pk.h2[0] = (_Float16)p[0];
                pk.h2[1] = (_Float16)p[kH * kW];
                packed = pk.u;
            }
            sB[(row * COLS + col) * PSTR + pair] = packed;
        }
        __syncthreads();

        const int cb = cch + hi * 16;  // A-frag channel base: contiguous k order (k = 16*hi + s)
#pragma unroll
        for (int j = 0; j < 9; ++j) {
            const int dy = j / 3 - 1;
            const int dx = j - (j / 3) * 3 - 1;
            // B fragments from LDS: lanes 0-15 hold K=0..15 (pairs 0..7), 16-31 K=16..31
            v16h bfr[2];
#pragma unroll
            for (int nt = 0; nt < 2; ++nt) {
                const int col = nt * 16 + l15 + dx + 1;     // in [0,33]
                const uint32_t* src = &sB[((dy + 1) * COLS + col) * PSTR + hi * 8];
                union { v4u u[2]; v16h v; } ub;
                ub.u[0] = *(const v4u*)(src);
                ub.u[1] = *(const v4u*)(src + 4);
                bfr[nt] = ub.v;
            }
#pragma unroll
            for (int mt = 0; mt < 4; ++mt) {
                const _Float16* pa =
                    A_h + (((size_t)j * kEMB + (erow0 + mt * 16)) * kC + cb);
                v8h lo8 = *(const v8h*)(pa);
                v8h hi8 = *(const v8h*)(pa + 8);
                v16h afr = __builtin_shufflevector(lo8, hi8,
                    0, 1, 2, 3, 4, 5, 6, 7, 8, 9, 10, 11, 12, 13, 14, 15);
                acc[mt][0] = __builtin_amdgcn_wmma_f32_16x16x32_f16(
                    false, afr, false, bfr[0], (short)0, acc[mt][0], false, false);
                acc[mt][1] = __builtin_amdgcn_wmma_f32_16x16x32_f16(
                    false, afr, false, bfr[1], (short)0, acc[mt][1], false, false);
            }
        }
    }

    // fused epilogue: e[l] = sum_e w_att[e]*tanh(x_em[e,l] + b_conv[e] + g_em[e])
    float part0 = 0.f, part1 = 0.f;
#pragma unroll
    for (int mt = 0; mt < 4; ++mt) {
#pragma unroll
        for (int r = 0; r < 8; ++r) {
            const int erow = wave * 64 + mt * 16 + r + hi * 8;  // f32 C/D layout
            const float bg = sBG[erow];
            const float wa = sWA[erow];
            part0 += tanhf(acc[mt][0][r] + bg) * wa;
            part1 += tanhf(acc[mt][1][r] + bg) * wa;
        }
    }
    part0 += __shfl_xor(part0, 16);   // combine lane halves (same n, disjoint M rows)
    part1 += __shfl_xor(part1, 16);
    if (lane < 16) {
        atomicAdd(&sE[l15], part0);
        atomicAdd(&sE[16 + l15], part1);
    }
    __syncthreads();
    if (tid < 32)
        Escore[(size_t)bb * kL + tile * 32 + tid] = sE[tid] + b_att[0];
}

// ---------------- kernel 3: softmax over L + alpha-weighted sum of conv_f ----------------
__global__ void __launch_bounds__(256)
k_softmax_out(const float* __restrict__ Escore, const float* __restrict__ conv_f,
              float* __restrict__ att_out, float* __restrict__ alpha_out)
{
    __shared__ float sA[kL];
    __shared__ float red[256];
    const int tid = threadIdx.x;
    const int b   = blockIdx.x;

    float m = -1e30f;
    for (int i = tid; i < kL; i += 256) {
        float v = Escore[(size_t)b * kL + i];
        sA[i] = v;
        m = fmaxf(m, v);
    }
    red[tid] = m;
    __syncthreads();
    for (int s = 128; s > 0; s >>= 1) {
        if (tid < s) red[tid] = fmaxf(red[tid], red[tid + s]);
        __syncthreads();
    }
    const float mx = red[0];
    __syncthreads();

    float sum = 0.f;
    for (int i = tid; i < kL; i += 256) {
        float e = expf(sA[i] - mx);
        sA[i] = e;
        sum += e;
    }
    red[tid] = sum;
    __syncthreads();
    for (int s = 128; s > 0; s >>= 1) {
        if (tid < s) red[tid] += red[tid + s];
        __syncthreads();
    }
    const float inv = 1.0f / red[0];
    __syncthreads();

    for (int i = tid; i < kL; i += 256) {
        float a = sA[i] * inv;
        sA[i] = a;
        alpha_out[(size_t)b * kL + i] = a;
    }
    __syncthreads();

    // att_out[b,c] = sum_l alpha[l] * conv_f[b,c,l] ; wave per 64-channel slab
    const int lane = tid & 31;
    const int wave = tid >> 5;
    for (int c = wave * 64; c < wave * 64 + 64; ++c) {
        const float* p = conv_f + ((size_t)b * kC + c) * kL;
        float acc = 0.f;
        for (int i = lane; i < kL; i += 32) acc += sA[i] * p[i];
#pragma unroll
        for (int off = 16; off > 0; off >>= 1) acc += __shfl_xor(acc, off);
        if (lane == 0) att_out[(size_t)b * kC + c] = acc;
    }
}

// ---------------- launcher ----------------
extern "C" void kernel_launch(void* const* d_in, const int* in_sizes, int n_in,
                              void* d_out, int out_size, void* d_ws, size_t ws_size,
                              hipStream_t stream) {
    const float* conv_f = (const float*)d_in[0];
    const float* h      = (const float*)d_in[1];
    const float* W_h    = (const float*)d_in[2];
    const float* b_h    = (const float*)d_in[3];
    const float* K_conv = (const float*)d_in[4];
    const float* b_conv = (const float*)d_in[5];
    const float* w_att  = (const float*)d_in[6];
    const float* b_att  = (const float*)d_in[7];

    char* ws = (char*)d_ws;
    _Float16* A_h   = (_Float16*)ws;                              // 9*512*512*2 = 4718592 B
    float*    g_em  = (float*)(ws + 4718592);                     // 128*512*4   = 262144 B
    float*    E     = (float*)(ws + 4718592 + 262144);            // 128*512*4   = 262144 B

    float* att_out   = (float*)d_out;                             // [B, C]
    float* alpha_out = att_out + (size_t)kB * kC;                 // [B, L]

    k_prep_weights<<<(kEMB * kC * 9) / 256, 256, 0, stream>>>(K_conv, A_h);
    k_gem<<<kB, 512, 0, stream>>>(h, W_h, b_h, g_em);
    k_conv_att<<<kB * 16, 256, 0, stream>>>(conv_f, A_h, g_em, b_conv, w_att, b_att, E);
    k_softmax_out<<<kB, 256, 0, stream>>>(E, conv_f, att_out, alpha_out);
}
